// MashDecoder_4191888081331
// MI455X (gfx1250) — matrix operand order repaired and checked by reading, not compiled, not run on hardware
//
#include <hip/hip_runtime.h>
#include <math.h>

// ---------------------------------------------------------------------------
// MASH decoder for MI455X (gfx1250). All GEMMs use v_wmma_f32_16x16x32_bf16
// (wave32, one 32x32 output tile per wave, f32 accumulate, bf16 operands).
// ---------------------------------------------------------------------------

typedef __attribute__((ext_vector_type(16))) __bf16 v16bf;
typedef __attribute__((ext_vector_type(8)))  float  v8f;

#define D_HIDDEN 400
#define D_INNER  800
#define D_STATE  16
#define DT_RANK  25
#define NB       4
#define NA       400
#define NQ       4096
#define NHEAD    8
#define DHEAD    400
#define INNER    3200

__device__ __forceinline__ void pack4(v16bf& f, int o, float4 q)
{
    f[o + 0] = (__bf16)q.x;
    f[o + 1] = (__bf16)q.y;
    f[o + 2] = (__bf16)q.z;
    f[o + 3] = (__bf16)q.w;
}

// ---------------------------------------------------------------------------
// Generic WMMA GEMM:  C[M,N] = alpha * A[M,K] x op(B)  (+ bias[n]) (+ Cin)
//   TRANS_B = true : B is [N,K] row-major (y = x @ W^T, the torch Linear case)
//   TRANS_B = false: B is [K,N] row-major (used for attn @ V)
// One wave per 32x32 tile. M must be a multiple of 32 (all our Ms are
// 1600/4096/16384). N is handled by clamping load addresses (dead accumulator
// columns) and guarding only the stores. K tail is masked branchlessly with
// v_cndmask-style selects. All alignment facts hold by construction:
// every lda/ldb and base offset in this model is a multiple of 4 floats.
// A/B fragments use contiguous k = 16*half + e, not the ISA table order; f32 C/D per CDNA5 ISA 7.12.2.
// ---------------------------------------------------------------------------
template <bool TRANS_B>
__global__ void __launch_bounds__(32)
gemm_wmma_k(const float* __restrict__ A, const float* __restrict__ Bp,
            const float* __restrict__ bias, const float* __restrict__ Cin,
            float* __restrict__ C,
            int M, int N, int K, int lda, int ldb, int ldc, float alpha)
{
    const int lane = threadIdx.x & 31;
    const int half = lane >> 4;     // 0: lanes 0-15, 1: lanes 16-31
    const int l16  = lane & 15;
    const int m0 = blockIdx.y * 32;
    const int n0 = blockIdx.x * 32;

    // Per-lane A row pointers (include the half*16 K-offset of the layout).
    const float* Ar0 = A + (long long)(m0 + l16) * lda + half * 16;
    const float* Ar1 = Ar0 + (long long)16 * lda;

    // Column indices for the two B fragments; clamp for always-valid loads.
    const int nIdx[2] = { n0 + l16, n0 + 16 + l16 };
    const int nCl[2]  = { nIdx[0] < N ? nIdx[0] : N - 1,
                          nIdx[1] < N ? nIdx[1] : N - 1 };

    v8f acc[2][2] = {};

    const int kfull = K & ~31;

    for (int kb = 0; kb < kfull; kb += 32) {
        v16bf a[2], b[2];

        // ---- A fragments: one contiguous 16-float run -> float4 loads
        {
            const float4* p0 = (const float4*)(Ar0 + kb);
            const float4* p1 = (const float4*)(Ar1 + kb);
            pack4(a[0], 0, p0[0]); pack4(a[0], 4, p0[1]);   // k rel 0..7
            pack4(a[0], 8, p0[2]); pack4(a[0], 12, p0[3]);  // k rel 8..15
            pack4(a[1], 0, p1[0]); pack4(a[1], 4, p1[1]);
            pack4(a[1], 8, p1[2]); pack4(a[1], 12, p1[3]);
            __builtin_prefetch(Ar0 + kb + 32, 0, 0);
            __builtin_prefetch(Ar1 + kb + 32, 0, 0);
        }

        // ---- B fragments
        #pragma unroll
        for (int t = 0; t < 2; ++t) {
            if (TRANS_B) {
                const float* p = Bp + (long long)nCl[t] * ldb + kb + half * 16;
                const float4* p4 = (const float4*)p;
                pack4(b[t], 0, p4[0]); pack4(b[t], 4, p4[1]);
                pack4(b[t], 8, p4[2]); pack4(b[t], 12, p4[3]);
                __builtin_prefetch(p + 32, 0, 0);
            } else {
                const float* p = Bp + (long long)(kb + half * 16) * ldb + nCl[t];
                #pragma unroll
                for (int e = 0; e < 16; ++e)
                    b[t][e] = (__bf16)p[(long long)e * ldb];
            }
        }

        #pragma unroll
        for (int i = 0; i < 2; ++i)
            #pragma unroll
            for (int j = 0; j < 2; ++j)
                acc[i][j] = __builtin_amdgcn_wmma_f32_16x16x32_bf16(
                    false, a[i], false, b[j], (short)0, acc[i][j], false, false);
    }

    // ---- K tail (branchless masking: unconditional clamped load + select)
    if (kfull < K) {
        const int kb = kfull;
        v16bf a[2], b[2];
        #pragma unroll
        for (int t = 0; t < 2; ++t) {
            const float* base = A + (long long)(m0 + t * 16 + l16) * lda;
            #pragma unroll
            for (int e = 0; e < 16; ++e) {
                int k = kb + half * 16 + e;
                int kc = k < K ? k : K - 1;
                float v = base[kc];
                a[t][e] = (__bf16)(k < K ? v : 0.0f);
            }
        }
        #pragma unroll
        for (int t = 0; t < 2; ++t) {
            #pragma unroll
            for (int e = 0; e < 16; ++e) {
                int k = kb + half * 16 + e;
                int kc = k < K ? k : K - 1;
                float v = TRANS_B ? Bp[(long long)nCl[t] * ldb + kc]
                                  : Bp[(long long)kc * ldb + nCl[t]];
                b[t][e] = (__bf16)(k < K ? v : 0.0f);
            }
        }
        #pragma unroll
        for (int i = 0; i < 2; ++i)
            #pragma unroll
            for (int j = 0; j < 2; ++j)
                acc[i][j] = __builtin_amdgcn_wmma_f32_16x16x32_bf16(
                    false, a[i], false, b[j], (short)0, acc[i][j], false, false);
    }

    // ---- store: VGPR r holds row m = r + 8*half, col n = l16 (guarded)
    #pragma unroll
    for (int i = 0; i < 2; ++i) {
        #pragma unroll
        for (int j = 0; j < 2; ++j) {
            int n = nIdx[j];
            if (n >= N) continue;
            #pragma unroll
            for (int r = 0; r < 8; ++r) {
                int m = m0 + i * 16 + r + half * 8;
                float v = alpha * acc[i][j][r];
                if (bias) v += bias[n];
                if (Cin)  v += Cin[(long long)m * ldc + n];
                C[(long long)m * ldc + n] = v;
            }
        }
    }
}

// ---------------------------------------------------------------------------
// Fourier point embeds
// ---------------------------------------------------------------------------
__global__ void __launch_bounds__(128)
anchor_embed_k(const float* __restrict__ mash, const float* __restrict__ w,
               const float* __restrict__ b, float* __restrict__ x)
{
    int row = blockIdx.x;                       // 0 .. 1599  (b*400+a)
    const float* m = mash + (long long)row * 31;
    __shared__ float emb[54];
    int t = threadIdx.x;
    if (t < 24) {                               // d-major: proj[d*4+f]
        int d = t >> 2, f = t & 3;
        float pr = m[d] * (float)(1 << f) * 3.14159265358979f;
        emb[t]      = sinf(pr);
        emb[24 + t] = cosf(pr);
    } else if (t < 30) {
        emb[48 + (t - 24)] = m[t - 24];
    }
    __syncthreads();
    float* xr = x + (long long)row * D_HIDDEN;
    for (int j = threadIdx.x; j < D_HIDDEN; j += 128) {
        if (j < 375) {
            float s = b[j];
            const float* wr = w + j * 54;
            #pragma unroll 6
            for (int q = 0; q < 54; ++q) s += emb[q] * wr[q];
            xr[j] = s;
        } else {
            xr[j] = m[6 + (j - 375)];
        }
    }
}

__global__ void __launch_bounds__(128)
point_embed_k(const float* __restrict__ qry, const float* __restrict__ w,
              const float* __restrict__ b, float* __restrict__ x)
{
    int row = blockIdx.x;                       // 0 .. 16383 (b*4096+i)
    const float* m = qry + (long long)row * 3;
    __shared__ float emb[51];
    int t = threadIdx.x;
    if (t < 24) {                               // proj[d*8+f]
        int d = t >> 3, f = t & 7;
        float pr = m[d] * (float)(1 << f) * 3.14159265358979f;
        emb[t]      = sinf(pr);
        emb[24 + t] = cosf(pr);
    } else if (t < 27) {
        emb[48 + (t - 24)] = m[t - 24];
    }
    __syncthreads();
    float* xr = x + (long long)row * D_HIDDEN;
    for (int j = threadIdx.x; j < D_HIDDEN; j += 128) {
        float s = b[j];
        const float* wr = w + j * 51;
        #pragma unroll 3
        for (int q = 0; q < 51; ++q) s += emb[q] * wr[q];
        xr[j] = s;
    }
}

// ---------------------------------------------------------------------------
// Norms (one 128-thread block per row, width 400)
// ---------------------------------------------------------------------------
__global__ void __launch_bounds__(128)
rmsnorm_k(const float* __restrict__ x, const float* __restrict__ w,
          float* __restrict__ o, int width)
{
    int row = blockIdx.x;
    const float* xr = x + (long long)row * width;
    float s = 0.f;
    for (int i = threadIdx.x; i < width; i += 128) { float v = xr[i]; s += v * v; }
    __shared__ float buf[128];
    buf[threadIdx.x] = s; __syncthreads();
    for (int st = 64; st > 0; st >>= 1) {
        if (threadIdx.x < st) buf[threadIdx.x] += buf[threadIdx.x + st];
        __syncthreads();
    }
    float sc = rsqrtf(buf[0] / width + 1e-5f);
    float* orow = o + (long long)row * width;
    for (int i = threadIdx.x; i < width; i += 128) orow[i] = xr[i] * sc * w[i];
}

__global__ void __launch_bounds__(128)
layernorm_k(const float* __restrict__ x, const float* __restrict__ w,
            const float* __restrict__ b, float* __restrict__ o, int width)
{
    int row = blockIdx.x;
    const float* xr = x + (long long)row * width;
    float s = 0.f, s2 = 0.f;
    for (int i = threadIdx.x; i < width; i += 128) {
        float v = xr[i]; s += v; s2 += v * v;
    }
    __shared__ float b1[128], b2[128];
    b1[threadIdx.x] = s; b2[threadIdx.x] = s2; __syncthreads();
    for (int st = 64; st > 0; st >>= 1) {
        if (threadIdx.x < st) { b1[threadIdx.x] += b1[threadIdx.x + st];
                                b2[threadIdx.x] += b2[threadIdx.x + st]; }
        __syncthreads();
    }
    float mu  = b1[0] / width;
    float var = b2[0] / width - mu * mu;
    float sc  = rsqrtf(var + 1e-5f);
    float* orow = o + (long long)row * width;
    for (int i = threadIdx.x; i < width; i += 128)
        orow[i] = (xr[i] - mu) * sc * w[i] + b[i];
}

// ---------------------------------------------------------------------------
// Mamba pieces
// ---------------------------------------------------------------------------
__global__ void conv_silu_k(const float* __restrict__ uz,   // [1600,1600], u = cols 0..799
                            const float* __restrict__ cw,   // [800,4]
                            const float* __restrict__ cb,   // [800]
                            float* __restrict__ uc)         // [1600,800]
{
    int idx = blockIdx.x * blockDim.x + threadIdx.x;        // ((b*400+l)*800+c)
    if (idx >= NB * NA * D_INNER) return;
    int c = idx % D_INNER;
    int l = (idx / D_INNER) % NA;
    int b = idx / (D_INNER * NA);
    float s = cb[c];
    #pragma unroll
    for (int k = 0; k < 4; ++k) {
        int ls = l - 3 + k;
        if (ls >= 0)
            s += uz[(long long)(b * NA + ls) * 1600 + c] * cw[c * 4 + k];
    }
    uc[idx] = s / (1.f + expf(-s));                         // SiLU
}

__global__ void dtproj_k(const float* __restrict__ dbc,     // [1600,57]
                         const float* __restrict__ dtw,     // [800,25]
                         const float* __restrict__ dtb,     // [800]
                         float* __restrict__ dt)            // [1600,800]
{
    int idx = blockIdx.x * blockDim.x + threadIdx.x;
    if (idx >= NB * NA * D_INNER) return;
    int c = idx % D_INNER;
    int r = idx / D_INNER;
    const float* p = dbc + (long long)r * 57;
    float s = dtb[c];
    #pragma unroll 5
    for (int t = 0; t < DT_RANK; ++t) s += p[t] * dtw[c * DT_RANK + t];
    dt[idx] = (s > 20.f) ? s : log1pf(expf(s));             // softplus
}

// Sequential selective scan: one thread per (batch, channel); state in regs.
__global__ void __launch_bounds__(64)
scan_k(const float* __restrict__ uc, const float* __restrict__ dt,
       const float* __restrict__ dbc, const float* __restrict__ A_log,
       const float* __restrict__ Dp, const float* __restrict__ uz,
       float* __restrict__ y)
{
    int idx = blockIdx.x * blockDim.x + threadIdx.x;
    if (idx >= NB * D_INNER) return;
    int d = idx % D_INNER;
    int b = idx / D_INNER;
    float A[D_STATE], h[D_STATE];
    #pragma unroll
    for (int s = 0; s < D_STATE; ++s) {
        A[s] = -expf(A_log[d * D_STATE + s]);
        h[s] = 0.f;
    }
    float Dv = Dp[d];
    for (int l = 0; l < NA; ++l) {
        long long row = (long long)(b * NA + l);
        float dtv = dt[row * D_INNER + d];
        float uv  = uc[row * D_INNER + d];
        const float* p = dbc + row * 57;      // [dt(25) | B(16) | C(16)]
        float yv = 0.f;
        #pragma unroll
        for (int s = 0; s < D_STATE; ++s) {
            h[s] = expf(dtv * A[s]) * h[s] + dtv * uv * p[25 + s];
            yv += h[s] * p[41 + s];
        }
        float zv = uz[row * 1600 + D_INNER + d];
        float sz = zv / (1.f + expf(-zv));
        y[row * D_INNER + d] = (yv + uv * Dv) * sz;
    }
}

// ---------------------------------------------------------------------------
// Attention softmax (row width 400)
// ---------------------------------------------------------------------------
__global__ void __launch_bounds__(128)
softmax_k(float* __restrict__ sim, int width)
{
    int row = blockIdx.x;
    float* p = sim + (long long)row * width;
    __shared__ float buf[128];
    float mx = -1e30f;
    for (int i = threadIdx.x; i < width; i += 128) mx = fmaxf(mx, p[i]);
    buf[threadIdx.x] = mx; __syncthreads();
    for (int st = 64; st > 0; st >>= 1) {
        if (threadIdx.x < st) buf[threadIdx.x] = fmaxf(buf[threadIdx.x], buf[threadIdx.x + st]);
        __syncthreads();
    }
    mx = buf[0]; __syncthreads();
    float s = 0.f;
    for (int i = threadIdx.x; i < width; i += 128) {
        float e = expf(p[i] - mx); p[i] = e; s += e;
    }
    buf[threadIdx.x] = s; __syncthreads();
    for (int st = 64; st > 0; st >>= 1) {
        if (threadIdx.x < st) buf[threadIdx.x] += buf[threadIdx.x + st];
        __syncthreads();
    }
    float inv = 1.f / buf[0];
    for (int i = threadIdx.x; i < width; i += 128) p[i] *= inv;
}

// ---------------------------------------------------------------------------
// GEGLU (in place on [rows,3200]: h[:, :1600] = a * gelu_tanh(g))
// ---------------------------------------------------------------------------
__global__ void geglu_k(float* __restrict__ h, int rows)
{
    long long idx = (long long)blockIdx.x * blockDim.x + threadIdx.x;
    if (idx >= (long long)rows * 1600) return;
    long long r = idx / 1600;
    int c = (int)(idx % 1600);
    float a = h[r * 3200 + c];
    float g = h[r * 3200 + 1600 + c];
    float t = tanhf(0.7978845608028654f * (g + 0.044715f * g * g * g));
    h[r * 3200 + c] = a * 0.5f * g * (1.f + t);
}

// ---------------------------------------------------------------------------
// Occupancy head
// ---------------------------------------------------------------------------
__global__ void head_k(const float* __restrict__ lat, const float* __restrict__ hw,
                       const float* __restrict__ hb, float* __restrict__ out, int rows)
{
    int r = blockIdx.x * blockDim.x + threadIdx.x;
    if (r >= rows) return;
    const float* p = lat + (long long)r * D_HIDDEN;
    float s = hb[0];
    for (int i = 0; i < D_HIDDEN; ++i) s += p[i] * hw[i];
    out[r] = s;
}

// ---------------------------------------------------------------------------
// Host-side GEMM wrappers
// ---------------------------------------------------------------------------
static inline void gemmT(const float* A, const float* B, const float* bias,
                         const float* Cin, float* C, int M, int N, int K,
                         int lda, int ldb, int ldc, float alpha, hipStream_t s)
{
    dim3 g((N + 31) / 32, (M + 31) / 32);
    gemm_wmma_k<true><<<g, 32, 0, s>>>(A, B, bias, Cin, C, M, N, K, lda, ldb, ldc, alpha);
}
static inline void gemmN(const float* A, const float* B, const float* bias,
                         const float* Cin, float* C, int M, int N, int K,
                         int lda, int ldb, int ldc, float alpha, hipStream_t s)
{
    dim3 g((N + 31) / 32, (M + 31) / 32);
    gemm_wmma_k<false><<<g, 32, 0, s>>>(A, B, bias, Cin, C, M, N, K, lda, ldb, ldc, alpha);
}

extern "C" void kernel_launch(void* const* d_in, const int* in_sizes, int n_in,
                              void* d_out, int out_size, void* d_ws, size_t ws_size,
                              hipStream_t stream)
{
    const float* mash     = (const float*)d_in[0];
    const float* qry      = (const float*)d_in[1];
    const float* anchor_w = (const float*)d_in[2];
    const float* anchor_b = (const float*)d_in[3];
    const float* point_w  = (const float*)d_in[4];
    const float* point_b  = (const float*)d_in[5];
    const float* norm_w   = (const float*)d_in[6];
    const float* in_w     = (const float*)d_in[7];
    const float* conv_w   = (const float*)d_in[8];
    const float* conv_b   = (const float*)d_in[9];
    const float* xproj_w  = (const float*)d_in[10];
    const float* dt_w     = (const float*)d_in[11];
    const float* dt_b     = (const float*)d_in[12];
    const float* A_log    = (const float*)d_in[13];
    const float* Dp       = (const float*)d_in[14];
    const float* out_w    = (const float*)d_in[15];
    const float* ln_q_w   = (const float*)d_in[16];
    const float* ln_q_b   = (const float*)d_in[17];
    const float* ln_c_w   = (const float*)d_in[18];
    const float* ln_c_b   = (const float*)d_in[19];
    const float* to_q_w   = (const float*)d_in[20];
    const float* to_kv_w  = (const float*)d_in[21];
    const float* to_out_w = (const float*)d_in[22];
    const float* to_out_b = (const float*)d_in[23];
    const float* ff_ln_w  = (const float*)d_in[24];
    const float* ff_ln_b  = (const float*)d_in[25];
    const float* ff_w1    = (const float*)d_in[26];
    const float* ff_b1    = (const float*)d_in[27];
    const float* ff_w2    = (const float*)d_in[28];
    const float* ff_b2    = (const float*)d_in[29];
    const float* head_w   = (const float*)d_in[30];
    const float* head_b   = (const float*)d_in[31];

    // ---- workspace layout (floats). Heavily aliased: BIGQ holds q then the
    //      FF hidden; QEMB holds q_emb -> qn (in-place LN) -> FF ln output.
    float* ws   = (float*)d_ws;
    float* X    = ws;                        // [1600,400]  x, then ctx (in-place LN)
    float* QEMB = X    + 1600 * 400;         // [16384,400]
    float* LAT  = QEMB + 16384 * 400;        // [16384,400]
    float* KV   = LAT  + 16384 * 400;        // [1600,6400]
    float* BIGQ = KV   + 1600 * 6400;        // [16384,3200] q, later FF hidden
    float* SIM  = BIGQ + (long long)16384 * 3200; // [4096,400]
    float* TMP  = SIM  + 4096 * 400;         // [4096,400]
    float* XN   = TMP  + 4096 * 400;         // [1600,400]
    float* UZ   = XN   + 1600 * 400;         // [1600,1600]
    float* UC   = UZ   + 1600 * 1600;        // [1600,800]
    float* DBC  = UC   + 1600 * 800;         // [1600,57]
    float* DT   = DBC  + 1600 * 57;          // [1600,800]
    float* YB   = DT   + 1600 * 800;         // [1600,800]

    const int ROWS_A = NB * NA;              // 1600
    const int ROWS_Q = NB * NQ;              // 16384

    // ---- 1. anchor embed -> x
    anchor_embed_k<<<ROWS_A, 128, 0, stream>>>(mash, anchor_w, anchor_b, X);

    // ---- 2. 24 Mamba blocks
    for (int i = 0; i < 24; ++i) {
        rmsnorm_k<<<ROWS_A, 128, 0, stream>>>(X, norm_w + i * 400, XN, 400);
        // uz = xn @ in_w^T                [1600,400]x[1600,400]^T -> [1600,1600]
        gemmT(XN, in_w + (long long)i * 1600 * 400, nullptr, nullptr, UZ,
              ROWS_A, 1600, 400, 400, 400, 1600, 1.f, stream);
        conv_silu_k<<<(ROWS_A * D_INNER + 255) / 256, 256, 0, stream>>>(
            UZ, conv_w + i * 800 * 4, conv_b + i * 800, UC);
        // dbc = uc @ xproj^T              [1600,800]x[57,800]^T -> [1600,57]
        gemmT(UC, xproj_w + (long long)i * 57 * 800, nullptr, nullptr, DBC,
              ROWS_A, 57, 800, 800, 800, 57, 1.f, stream);
        dtproj_k<<<(ROWS_A * D_INNER + 255) / 256, 256, 0, stream>>>(
            DBC, dt_w + (long long)i * 800 * 25, dt_b + i * 800, DT);
        scan_k<<<(NB * D_INNER + 63) / 64, 64, 0, stream>>>(
            UC, DT, DBC, A_log + (long long)i * 800 * 16, Dp + i * 800, UZ, YB);
        // x += (y * silu(z)) @ out_w^T    [1600,800]x[400,800]^T -> [1600,400]
        gemmT(YB, out_w + (long long)i * 400 * 800, nullptr, X, X,
              ROWS_A, 400, 800, 800, 800, 400, 1.f, stream);
    }

    // ---- 3. query embed + pre-norms (both LNs in place)
    point_embed_k<<<ROWS_Q, 128, 0, stream>>>(qry, point_w, point_b, QEMB);
    layernorm_k<<<ROWS_Q, 128, 0, stream>>>(QEMB, ln_q_w, ln_q_b, QEMB, 400);
    layernorm_k<<<ROWS_A, 128, 0, stream>>>(X, ln_c_w, ln_c_b, X, 400);

    // ---- 4. projections
    // q  = qn  @ to_q_w^T  -> [16384,3200]
    gemmT(QEMB, to_q_w, nullptr, nullptr, BIGQ,
          ROWS_Q, INNER, 400, 400, 400, INNER, 1.f, stream);
    // kv = ctx @ to_kv_w^T -> [1600,6400]
    gemmT(X, to_kv_w, nullptr, nullptr, KV,
          ROWS_A, 2 * INNER, 400, 400, 400, 2 * INNER, 1.f, stream);

    // ---- 5. cross attention, per (batch, head); accumulate to_out per head
    const float scale = 0.05f;               // 400^-0.5
    for (int bh = 0; bh < NB * NHEAD; ++bh) {
        int b = bh / NHEAD, h = bh % NHEAD;
        const float* Aq = BIGQ + (long long)b * NQ * INNER + h * DHEAD;
        const float* Bk = KV + (long long)b * NA * (2 * INNER) + h * DHEAD;
        // sim = q @ k^T * scale            [4096,400]
        gemmT(Aq, Bk, nullptr, nullptr, SIM,
              NQ, NA, DHEAD, INNER, 2 * INNER, NA, scale, stream);
        softmax_k<<<NQ, 128, 0, stream>>>(SIM, NA);
        // tmp = attn @ v                   [4096,400]  (B is [K=na, N=dh] strided)
        const float* Bv = KV + (long long)b * NA * (2 * INNER) + INNER + h * DHEAD;
        gemmN(SIM, Bv, nullptr, nullptr, TMP,
              NQ, DHEAD, NA, NA, 2 * INNER, DHEAD, 1.f, stream);
        // latents[b] (+)= tmp @ to_out_w[:, h*400:(h+1)*400]^T   (+bias at h==0)
        float* Cl = LAT + (long long)b * NQ * 400;
        gemmT(TMP, to_out_w + h * DHEAD,
              (h == 0) ? to_out_b : nullptr,
              (h == 0) ? nullptr : Cl, Cl,
              NQ, 400, DHEAD, DHEAD, INNER, 400, 1.f, stream);
    }

    // ---- 6. GEGLU FF with residual (QEMB reused as LN output, BIGQ as hidden)
    layernorm_k<<<ROWS_Q, 128, 0, stream>>>(LAT, ff_ln_w, ff_ln_b, QEMB, 400);
    gemmT(QEMB, ff_w1, ff_b1, nullptr, BIGQ,
          ROWS_Q, INNER, 400, 400, 400, INNER, 1.f, stream);
    geglu_k<<<(int)(((long long)ROWS_Q * 1600 + 255) / 256), 256, 0, stream>>>(BIGQ, ROWS_Q);
    gemmT(BIGQ, ff_w2, ff_b2, LAT, LAT,
          ROWS_Q, 400, 1600, INNER, 1600, 400, 1.f, stream);

    // ---- 7. occupancy head -> d_out [4,4096] f32
    head_k<<<(ROWS_Q + 255) / 256, 256, 0, stream>>>(
        LAT, head_w, head_b, (float*)d_out, ROWS_Q);
}
